// RNNLayer_15822659518949
// MI455X (gfx1250) — matrix operand order repaired and checked by reading, not compiled, not run on hardware
//
#include <hip/hip_runtime.h>
#include <hip/hip_bf16.h>

// ---------------- problem constants ----------------
#define T_   512
#define B_   256
#define D_   256
#define H_   512
#define G4_  2048            // gacc cols: [r(512) | z(512) | ghn(512) | gxn(512)]
#define NJB_H 16             // H/32 K-blocks for h fragments
#define NJB_X 8              // D/32 K-blocks for x fragments
#define NGT_  96             // 1536/16 column tiles per weight matrix

typedef __attribute__((ext_vector_type(16))) __bf16 v16bf;
typedef __attribute__((ext_vector_type(8)))  float  v8f;

#define WMMA_BF16(a, b, c) \
  __builtin_amdgcn_wmma_f32_16x16x32_bf16(false, (a), false, (b), (short)0, (c), false, false)

// fp32 -> bf16 round-to-nearest-even (raw bits)
__device__ __forceinline__ unsigned short f2bf(float x) {
  unsigned u = __float_as_uint(x);
  unsigned r = u + 0x7FFFu + ((u >> 16) & 1u);
  if ((u & 0x7F800000u) == 0x7F800000u) r = u;   // Inf/NaN: truncate
  return (unsigned short)(r >> 16);
}

// Scatter one element into a 16-bit A fragment (16x32, contiguous k = 16*half + i order).
__device__ __forceinline__ void store_afrag(unsigned short* frag, int NKB,
                                            int row, int k, float v) {
  int bt = row >> 4, m = row & 15;
  int jb = k >> 5;  int kk = k & 31;
  int lh = (kk >> 4) & 1;                       // lane half
  int i  = kk & 15;                             // bf16 index within lane
  int lane = lh * 16 + m;
  frag[(((bt * NKB + jb) * 32) + lane) * 16 + i] = f2bf(v);
}

// ---------------- weight packing (once per call) ----------------
__global__ __launch_bounds__(256) void pack_weights(
    const float* __restrict__ W_hh, const float* __restrict__ W_ih,
    unsigned short* __restrict__ whh, unsigned short* __restrict__ wih) {
  int tid  = blockIdx.x * 256 + threadIdx.x;    // 73728 threads
  int f    = tid >> 5, lane = tid & 31;
  int n    = lane & 15, klb = (lane >> 4) * 16;
  if (f < NGT_ * NJB_H) {                       // W_hh: 96 gt x 16 kb
    int gt = f >> 4, kb = f & 15;
    const float* src = W_hh + (gt * 16 + n) * H_ + kb * 32 + klb;
    unsigned short* dst = whh + (f * 32 + lane) * 16;
#pragma unroll
    for (int i = 0; i < 16; ++i) dst[i] = f2bf(src[i]);
  } else {                                      // W_ih: 96 gt x 8 kb
    int f2 = f - NGT_ * NJB_H;
    int gt = f2 >> 3, kb = f2 & 7;
    const float* src = W_ih + (gt * 16 + n) * D_ + kb * 32 + klb;
    unsigned short* dst = wih + (f2 * 32 + lane) * 16;
#pragma unroll
    for (int i = 0; i < 16; ++i) dst[i] = f2bf(src[i]);
  }
}

// ---------------- step 0 state prep ----------------
__global__ __launch_bounds__(256) void gru_prep(
    const float* __restrict__ hxs, const float* __restrict__ masks,
    const float* __restrict__ x, float* __restrict__ h_cur,
    unsigned short* __restrict__ hfrag, unsigned short* __restrict__ xfrag) {
  int idx = blockIdx.x * 256 + threadIdx.x;     // B*H threads
  int b = idx >> 9, n = idx & 511;
  float hm = hxs[b * H_ + n] * masks[b];        // mask t=0
  h_cur[idx] = hm;
  store_afrag(hfrag, NJB_H, b, n, hm);
  if (n < D_) store_afrag(xfrag, NJB_X, b, n, x[b * D_ + n]);  // x_0
}

// One K-chunk = 8 WMMA blocks. All 16 fragments are loaded into distinct,
// simultaneously-live registers (one clause of 32 global_load_b128), then the
// 8 chained WMMAs issue: the first WMMA's wait drains while later loads are
// still in flight -> ~1 L2 round trip per chunk instead of per WMMA.
template <int NKB>
__device__ __forceinline__ void gemm_seg1(const v16bf* __restrict__ ap,
                                          const v16bf* __restrict__ bp,
                                          v8f& c) {
#pragma unroll
  for (int base = 0; base < NKB; base += 8) {
    v16bf a0 = ap[(base + 0) * 32], a1 = ap[(base + 1) * 32];
    v16bf a2 = ap[(base + 2) * 32], a3 = ap[(base + 3) * 32];
    v16bf a4 = ap[(base + 4) * 32], a5 = ap[(base + 5) * 32];
    v16bf a6 = ap[(base + 6) * 32], a7 = ap[(base + 7) * 32];
    v16bf b0 = bp[(base + 0) * 32], b1 = bp[(base + 1) * 32];
    v16bf b2 = bp[(base + 2) * 32], b3 = bp[(base + 3) * 32];
    v16bf b4 = bp[(base + 4) * 32], b5 = bp[(base + 5) * 32];
    v16bf b6 = bp[(base + 6) * 32], b7 = bp[(base + 7) * 32];
    c = WMMA_BF16(a0, b0, c);
    c = WMMA_BF16(a1, b1, c);
    c = WMMA_BF16(a2, b2, c);
    c = WMMA_BF16(a3, b3, c);
    c = WMMA_BF16(a4, b4, c);
    c = WMMA_BF16(a5, b5, c);
    c = WMMA_BF16(a6, b6, c);
    c = WMMA_BF16(a7, b7, c);
  }
}

// ---------------- per-step GEMM: gacc = [A_h|A_x] x [W packs] ----------------
// 2048 waves, one 16x16 tile each: bt = wave&15, gq = wave>>4 in [0,128).
// gq <  64 : fused r/z  (h*Whh[gq] + x*Wih[gq])
// 64 -  95 : ghn        (h*Whh[gq])
// 96 - 127 : gxn        (x*Wih[gq-32])
__global__ __launch_bounds__(256) void gru_gemm(
    const v16bf* __restrict__ hA, const v16bf* __restrict__ xA,
    const v16bf* __restrict__ whh, const v16bf* __restrict__ wih,
    float* __restrict__ gacc) {
  const int wave = (blockIdx.x * 256 + threadIdx.x) >> 5;
  const int lane = threadIdx.x & 31;
  const int bt = wave & 15;
  const int gq = wave >> 4;
  v8f c = {};

  if (gq < 96) {                                // h x W_hh segment, K = 512
    gemm_seg1<NJB_H>(hA + bt * NJB_H * 32 + lane,
                     whh + gq * NJB_H * 32 + lane, c);
  }
  if (gq < 64 || gq >= 96) {                    // x x W_ih segment, K = 256
    const int gt = (gq < 64) ? gq : gq - 32;
    gemm_seg1<NJB_X>(xA + bt * NJB_X * 32 + lane,
                     wih + gt * NJB_X * 32 + lane, c);
  }
  // C/D 16x16 f32 layout: n = lane%16, m = v + 8*(lane/16)
  const int b0 = bt * 16 + (lane >> 4) * 8;
  const int g0 = gq * 16 + (lane & 15);
  float* dst = gacc + b0 * G4_ + g0;
#pragma unroll
  for (int v = 0; v < 8; ++v) dst[v * G4_] = c[v];
}

// ---------------- per-step gates + next-step state prep ----------------
__global__ __launch_bounds__(256) void gru_gates(
    const float* __restrict__ gacc, const float* __restrict__ b_ih,
    const float* __restrict__ b_hh, const float* __restrict__ masks,
    const float* __restrict__ x, float* __restrict__ h_cur,
    unsigned short* __restrict__ hfrag, unsigned short* __restrict__ xfrag,
    float* __restrict__ outRaw, float* __restrict__ hxs_out, int t) {
  int idx = blockIdx.x * 256 + threadIdx.x;     // B*H threads
  int b = idx >> 9, n = idx & 511;
  const float* g = gacc + b * G4_;
  float r  = 1.f / (1.f + __expf(-(g[n]        + b_ih[n]        + b_hh[n])));
  float z  = 1.f / (1.f + __expf(-(g[512 + n]  + b_ih[512 + n]  + b_hh[512 + n])));
  float ng = tanhf((g[1536 + n] + b_ih[1024 + n]) +
                   r * (g[1024 + n] + b_hh[1024 + n]));
  float h_new = (1.f - z) * ng + z * h_cur[idx];
  outRaw[(size_t)(t * B_ + b) * H_ + n] = h_new;
  if (t + 1 < T_) {
    float hm = h_new * masks[(t + 1) * B_ + b];
    h_cur[idx] = hm;
    store_afrag(hfrag, NJB_H, b, n, hm);
    if (n < D_)
      store_afrag(xfrag, NJB_X, b, n, x[(size_t)((t + 1) * B_ + b) * D_ + n]);
  } else {
    hxs_out[idx] = h_new;                       // final hidden state (unmasked)
  }
}

// ---------------- layernorm + cascade (one row per block) ----------------
__global__ __launch_bounds__(256) void ln_cascade(
    const float* __restrict__ raw, const float* __restrict__ prev,
    const float* __restrict__ rate_p, const float* __restrict__ gamma,
    const float* __restrict__ beta, float* __restrict__ normed,
    float* __restrict__ cascade) {
  __shared__ float s_sum[8], s_sq[8];
  const int tid = threadIdx.x;
  const size_t base = (size_t)blockIdx.x * H_;
  float o0 = raw[base + tid], o1 = raw[base + tid + 256];
  float s = o0 + o1, q = o0 * o0 + o1 * o1;
#pragma unroll
  for (int off = 16; off > 0; off >>= 1) {      // wave32 reduction
    s += __shfl_xor(s, off, 32);
    q += __shfl_xor(q, off, 32);
  }
  if ((tid & 31) == 0) { s_sum[tid >> 5] = s; s_sq[tid >> 5] = q; }
  __syncthreads();
  float ts = 0.f, tq = 0.f;
#pragma unroll
  for (int i = 0; i < 8; ++i) { ts += s_sum[i]; tq += s_sq[i]; }
  float mu   = ts * (1.f / H_);
  float var  = tq * (1.f / H_) - mu * mu;
  float rstd = rsqrtf(var + 1e-5f);
  float rate = *rate_p;
  cascade[base + tid]       = rate * o0 + (1.f - rate) * prev[base + tid];
  cascade[base + tid + 256] = rate * o1 + (1.f - rate) * prev[base + tid + 256];
  normed[base + tid]        = (o0 - mu) * rstd * gamma[tid]       + beta[tid];
  normed[base + tid + 256]  = (o1 - mu) * rstd * gamma[tid + 256] + beta[tid + 256];
}

// ---------------- host ----------------
extern "C" void kernel_launch(void* const* d_in, const int* in_sizes, int n_in,
                              void* d_out, int out_size, void* d_ws, size_t ws_size,
                              hipStream_t stream) {
  const float* x      = (const float*)d_in[0];
  const float* hxs    = (const float*)d_in[1];
  const float* masks  = (const float*)d_in[2];
  const float* prev   = (const float*)d_in[3];
  const float* rate   = (const float*)d_in[4];
  const float* W_ih   = (const float*)d_in[5];
  const float* W_hh   = (const float*)d_in[6];
  const float* b_ih   = (const float*)d_in[7];
  const float* b_hh   = (const float*)d_in[8];
  const float* gamma  = (const float*)d_in[9];
  const float* beta   = (const float*)d_in[10];

  float* out = (float*)d_out;
  float* outNormed  = out;                               // T*B*H
  float* outHxs     = out + (size_t)T_ * B_ * H_;        // B*H
  float* outCascade = outHxs + (size_t)B_ * H_;          // T*B*H

  char* ws = (char*)d_ws;
  const size_t OFF_WHH   = 0;                            // 1,572,864 B
  const size_t OFF_WIH   = 1572864;                      //   786,432 B
  const size_t OFF_HFRAG = 2359296;                      //   262,144 B
  const size_t OFF_XFRAG = 2621440;                      //   131,072 B
  const size_t OFF_HCUR  = 2752512;                      //   524,288 B
  const size_t OFF_GACC  = 3276800;                      // 2,097,152 B -> 5.4 MB total
  unsigned short* whhP = (unsigned short*)(ws + OFF_WHH);
  unsigned short* wihP = (unsigned short*)(ws + OFF_WIH);
  unsigned short* hfr  = (unsigned short*)(ws + OFF_HFRAG);
  unsigned short* xfr  = (unsigned short*)(ws + OFF_XFRAG);
  float* h_cur = (float*)(ws + OFF_HCUR);
  float* gacc  = (float*)(ws + OFF_GACC);

  pack_weights<<<288, 256, 0, stream>>>(W_hh, W_ih, whhP, wihP);
  gru_prep<<<512, 256, 0, stream>>>(hxs, masks, x, h_cur, hfr, xfr);

  for (int t = 0; t < T_; ++t) {
    gru_gemm<<<256, 256, 0, stream>>>((const v16bf*)hfr, (const v16bf*)xfr,
                                      (const v16bf*)whhP, (const v16bf*)wihP, gacc);
    gru_gates<<<512, 256, 0, stream>>>(gacc, b_ih, b_hh, masks, x, h_cur,
                                       hfr, xfr, outNormed, outHxs, t);
  }
  ln_cascade<<<T_ * B_, 256, 0, stream>>>(outNormed, prev, rate, gamma, beta,
                                          outNormed, outCascade);
}
